// CausalI_47132971106806
// MI455X (gfx1250) — matrix operand order repaired and checked by reading, not compiled, not run on hardware
//
#include <hip/hip_runtime.h>

#define H 128
#define EPSV 1e-5f
#define BN_BETA 1e-4f

typedef __attribute__((ext_vector_type(16))) __bf16 v16bf;
typedef __attribute__((ext_vector_type(8)))  __bf16 v8bf;
typedef __attribute__((ext_vector_type(8)))  float  v8f;

static __device__ __forceinline__ unsigned short f2bf(float f) {
    unsigned int u = __float_as_uint(f);
    u += 0x7FFFu + ((u >> 16) & 1u);   // round-to-nearest-even
    return (unsigned short)(u >> 16);
}

// ---------------- utility kernels ----------------
__global__ void k_fill(float* p, long n, float v) {
    long i = (long)blockIdx.x * blockDim.x + threadIdx.x;
    if (i < n) p[i] = v;
}

// standalone per-column stats: val[n,c] = src[(perm?perm[n]:n),c] (+ src2[n,c])
__global__ void k_colstats(const float* __restrict__ src, const float* __restrict__ src2,
                           const int* __restrict__ perm, float* __restrict__ statsOut,
                           int N, int rowsPerBlock) {
    int c  = threadIdx.x;                    // 0..127
    int r0 = blockIdx.x * rowsPerBlock;
    int r1 = r0 + rowsPerBlock; if (r1 > N) r1 = N;
    float s = 0.f, q = 0.f;
    for (int n = r0; n < r1; ++n) {
        int idx = perm ? perm[n] : n;
        float v = src[(long)idx * H + c];
        if (src2) v += src2[(long)n * H + c];
        s += v; q += v * v;
    }
    atomicAdd(&statsOut[c], s);
    atomicAdd(&statsOut[H + c], q);
}

__global__ void k_stats_fin(const float* statsOut, float* mu, float* rstd, int N) {
    int c = threadIdx.x;
    float m = statsOut[c] / (float)N;
    float v = statsOut[H + c] / (float)N - m * m;  // biased variance (jnp.var)
    mu[c]   = m;
    rstd[c] = rsqrtf(v + EPSV);
}

// standalone fp32 -> bf16 (used for raw input x and the perm-gathered xc[perm]+xo)
__global__ void k_cvt(const float* __restrict__ src, const float* __restrict__ src2,
                      const int* __restrict__ perm, unsigned short* __restrict__ dst, int N) {
    long i = (long)blockIdx.x * blockDim.x + threadIdx.x;
    if (i >= (long)N * H) return;
    int n = (int)(i >> 7), c = (int)(i & (H - 1));
    int idx = perm ? perm[n] : n;
    float v = src[(long)idx * H + c];
    if (src2) v += src2[i];
    dst[i] = f2bf(v);
}

// Fold BN into weights: WT[j,k] = bf16(rstd[k]*W[k,j]);
// biasp[j] = b[j] + sum_k (beta - mu[k]*rstd[k]) * W[k,j].  W is [128, nCols] row-major.
__global__ void k_fold(const float* __restrict__ W, const float* __restrict__ b,
                       const float* __restrict__ mu, const float* __restrict__ rstd,
                       unsigned short* __restrict__ WT, float* __restrict__ biasp,
                       int nCols, int padCols) {
    int j = threadIdx.x;
    if (j >= padCols) return;
    if (j >= nCols) {
        for (int k = 0; k < H; ++k) WT[j * H + k] = 0;
        if (j < H) biasp[j] = 0.f;
        return;
    }
    float acc = b ? b[j] : 0.f;
    for (int k = 0; k < H; ++k) {
        float w  = W[k * nCols + j];
        float rs = rstd[k];
        WT[j * H + k] = f2bf(w * rs);
        acc += (BN_BETA - mu[k] * rs) * w;
    }
    biasp[j] = acc;
}

// Pack [W_ea_top(2) | W_ea_bot(2) | W_na(2) | zeros] transposed into 32x128 bf16 (padded).
__global__ void k_fold_att(const float* __restrict__ W_ea, const float* __restrict__ W_na,
                           const float* __restrict__ b_na,
                           unsigned short* __restrict__ WT, float* __restrict__ biasp) {
    int t = blockIdx.x * blockDim.x + threadIdx.x;
    if (t >= 32 * H) return;
    int j = t / H, k = t % H;
    float w = 0.f;
    if (j < 2)      w = W_ea[k * 2 + j];             // u = h . W_ea[0:128]
    else if (j < 4) w = W_ea[(H + k) * 2 + (j - 2)]; // v = h . W_ea[128:256]
    else if (j < 6) w = W_na[k * 2 + (j - 4)];       // node-att logits
    WT[t] = f2bf(w);
    if (k == 0) biasp[j] = (j == 4 || j == 5) ? b_na[j - 4] : 0.f;
}

// ------------- WMMA GEMM: out[N, ldOut] = A[N,128] @ WT^T + biasp -------------
// One wave computes a 16x32 output tile (1 A-frag reused for 2 B tiles, 8 WMMAs).
// Optional fused epilogue: ReLU, bf16 copy (stride H), per-column BN stats.
__global__ __launch_bounds__(256) void k_gemm(
        const unsigned short* __restrict__ A, const unsigned short* __restrict__ WT,
        const float* __restrict__ biasp, float* __restrict__ Cout,
        unsigned short* __restrict__ bfOut, float* __restrict__ statsOut,
        int N, int ldOut, int tilesN, int relu) {
    int lane  = threadIdx.x & 31;
    int wid   = threadIdx.x >> 5;
    int pairs = (tilesN + 1) >> 1;
    int gw    = blockIdx.x * (blockDim.x >> 5) + wid;
    int tilesM = (N + 15) >> 4;
    if (gw >= tilesM * pairs) return;
    int tm = gw / pairs, tn0 = (gw % pairs) * 2;
    int hi = lane >> 4;          // lane half
    int lm = lane & 15;
    int arow = tm * 16 + lm; if (arow >= N) arow = N - 1;
    const unsigned short* pa  = A  + (long)arow * H;
    const unsigned short* pb0 = WT + (long)(tn0 * 16 + lm) * H;
    const unsigned short* pb1 = pb0 + 16 * H;  // padded rows exist; result discarded if unused

    v8f c0 = {}, c1 = {};
#pragma unroll
    for (int kb = 0; kb < 4; ++kb) {
        // A 16x32 frag: lane=M; elems 0..7 <- K=kb*32+hi*8, elems 8..15 <- +16
        int k0a = kb * 32 + hi * 8;
        v8bf a0 = *(const v8bf*)(pa + k0a);
        v8bf a1 = *(const v8bf*)(pa + k0a + 16);
        // B 32x16 frag: lane=N-col; elems 0..7 <- K=kb*32+hi*8, elems 8..15 <- +16
        int k0b = kb * 32 + hi * 8;
        v8bf b00 = *(const v8bf*)(pb0 + k0b);
        v8bf b01 = *(const v8bf*)(pb0 + k0b + 16);
        v8bf b10 = *(const v8bf*)(pb1 + k0b);
        v8bf b11 = *(const v8bf*)(pb1 + k0b + 16);
        v16bf a, b0, b1;
#pragma unroll
        for (int i = 0; i < 8; ++i) {
            a[i]  = a0[i];  a[8 + i]  = a1[i];
            b0[i] = b00[i]; b0[8 + i] = b01[i];
            b1[i] = b10[i]; b1[8 + i] = b11[i];
        }
        c0 = __builtin_amdgcn_wmma_f32_16x16x32_bf16(false, a, false, b0, (short)0, c0, false, false);
        c1 = __builtin_amdgcn_wmma_f32_16x16x32_bf16(false, a, false, b1, (short)0, c1, false, false);
    }

    bool full = (tm * 16 + 16 <= N);
    auto storeTile = [&](v8f c, int tn) {
        int col = tn * 16 + lm;
        if (col >= ldOut) return;
        float bj = biasp[col];
        float s = 0.f, q = 0.f;
        long base = (long)(tm * 16) * ldOut + col;
        if (full) {
#pragma unroll
            for (int r = 0; r < 8; ++r) {   // c[r] = D[M = r + 8*hi][col]
                float v = c[r] + bj;
                if (relu) v = fmaxf(v, 0.f);
                long rowi = r + 8 * hi;
                Cout[base + rowi * ldOut] = v;
                if (bfOut) bfOut[(long)(tm * 16 + rowi) * H + col] = f2bf(v);
                s += v; q += v * v;
            }
        } else {
            for (int r = 0; r < 8; ++r) {
                int rowi = tm * 16 + r + 8 * hi;
                if (rowi >= N) continue;
                float v = c[r] + bj;
                if (relu) v = fmaxf(v, 0.f);
                Cout[(long)rowi * ldOut + col] = v;
                if (bfOut) bfOut[(long)rowi * H + col] = f2bf(v);
                s += v; q += v * v;
            }
        }
        if (statsOut) {
            atomicAdd(&statsOut[col], s);
            atomicAdd(&statsOut[H + col], q);
        }
    };
    storeTile(c0, tn0);
    if (tn0 + 1 < tilesN) storeTile(c1, tn0 + 1);
}

// ---------------- graph kernels ----------------
__global__ void k_deg_edges(const int* __restrict__ rows, float* __restrict__ deg, int E) {
    int e = blockIdx.x * blockDim.x + threadIdx.x;
    if (e >= E) return;
    atomicAdd(&deg[rows[e]], 1.f);
}

__global__ void k_rsqrt(float* p, int n) {
    int i = blockIdx.x * blockDim.x + threadIdx.x;
    if (i < n) p[i] = rsqrtf(p[i]);
}

// self-loop: agg[n,:] = t[n,:] * dinv[n]^2
__global__ void k_selfloop(const float* __restrict__ t, const float* __restrict__ dinv,
                           float* __restrict__ agg, int N) {
    long i = (long)blockIdx.x * blockDim.x + threadIdx.x;
    if (i >= (long)N * H) return;
    float d = dinv[i >> 7];
    agg[i] = t[i] * d * d;
}

// one wave per edge (32 lanes x float4 = full 128-f row): L2-resident float atomics
__global__ void k_scatter(const float* __restrict__ t, float* __restrict__ agg,
                          const int* __restrict__ rows, const int* __restrict__ cols,
                          const float* __restrict__ dinv, const float* __restrict__ ew2,
                          int comp, int E) {
    long tid = (long)blockIdx.x * blockDim.x + threadIdx.x;
    int e = (int)(tid >> 5), g = (int)(tid & 31);
    if (e >= E) return;
    int r = rows[e], c = cols[e];
    float w = ew2 ? ew2[2 * e + comp] : 1.f;
    float coef = dinv[r] * w * dinv[c];
    float4 v = *(const float4*)(t + (long)r * H + g * 4);
    float* dst = agg + (long)c * H + g * 4;
    atomicAdd(dst + 0, v.x * coef);
    atomicAdd(dst + 1, v.y * coef);
    atomicAdd(dst + 2, v.z * coef);
    atomicAdd(dst + 3, v.w * coef);
}

// fused: out = relu(agg + b); bf16 copy; per-column stats of out.  128 threads (col).
__global__ void k_bias_relu(const float* __restrict__ agg, const float* __restrict__ b,
                            float* __restrict__ out, unsigned short* __restrict__ bfOut,
                            float* __restrict__ statsOut, int N, int rowsPerBlock) {
    int c = threadIdx.x;
    float bc = b[c];
    int r0 = blockIdx.x * rowsPerBlock;
    int r1 = r0 + rowsPerBlock; if (r1 > N) r1 = N;
    float s = 0.f, q = 0.f;
    for (int n = r0; n < r1; ++n) {
        long i = (long)n * H + c;
        float v = fmaxf(agg[i] + bc, 0.f);
        out[i]   = v;
        bfOut[i] = f2bf(v);
        s += v; q += v * v;
    }
    atomicAdd(&statsOut[c], s);
    atomicAdd(&statsOut[H + c], q);
}

// edge attention softmax over 2 logits (u[row]+v[col]+b_ea) + weighted-degree accumulation
__global__ void k_edge_att(const float* __restrict__ att, const int* __restrict__ rows,
                           const int* __restrict__ cols, const float* __restrict__ b_ea,
                           float* __restrict__ ea, float* __restrict__ degc,
                           float* __restrict__ dego, int E) {
    int e = blockIdx.x * blockDim.x + threadIdx.x;
    if (e >= E) return;
    int r = rows[e], c = cols[e];
    float l0 = att[(long)r * 16 + 0] + att[(long)c * 16 + 2] + b_ea[0];
    float l1 = att[(long)r * 16 + 1] + att[(long)c * 16 + 3] + b_ea[1];
    float m = fmaxf(l0, l1);
    float e0 = __expf(l0 - m), e1 = __expf(l1 - m);
    float s = e0 + e1; e0 /= s; e1 /= s;
    ea[2 * e] = e0; ea[2 * e + 1] = e1;
    atomicAdd(&degc[r], e0);
    atomicAdd(&dego[r], e1);
}

// fused node attention: xc = na0*h, xo = na1*h; bf16 copies; stats for both streams.
__global__ void k_node_att(const float* __restrict__ att, const float* __restrict__ h,
                           float* __restrict__ xc, float* __restrict__ xo,
                           unsigned short* __restrict__ xcb, unsigned short* __restrict__ xob,
                           float* __restrict__ sA, float* __restrict__ sB,
                           int N, int rowsPerBlock) {
    int c = threadIdx.x;
    int r0 = blockIdx.x * rowsPerBlock;
    int r1 = r0 + rowsPerBlock; if (r1 > N) r1 = N;
    float sc = 0.f, qc = 0.f, so = 0.f, qo = 0.f;
    for (int n = r0; n < r1; ++n) {
        float l0 = att[(long)n * 16 + 4], l1 = att[(long)n * 16 + 5];
        float m = fmaxf(l0, l1);
        float e0 = __expf(l0 - m), e1 = __expf(l1 - m);
        float inv = 1.f / (e0 + e1);
        long i = (long)n * H + c;
        float hv = h[i];
        float vc = hv * (e0 * inv), vo = hv * (e1 * inv);
        xc[i] = vc; xcb[i] = f2bf(vc); sc += vc; qc += vc * vc;
        xo[i] = vo; xob[i] = f2bf(vo); so += vo; qo += vo * vo;
    }
    atomicAdd(&sA[c], sc); atomicAdd(&sA[H + c], qc);
    atomicAdd(&sB[c], so); atomicAdd(&sB[H + c], qo);
}

__global__ void k_logsoftmax(const float* __restrict__ logits, float* __restrict__ out,
                             int N, int C) {
    int n = blockIdx.x * blockDim.x + threadIdx.x;
    if (n >= N) return;
    const float* p = logits + (long)n * 16;
    float m = -1e30f;
    for (int c = 0; c < C; ++c) m = fmaxf(m, p[c]);
    float s = 0.f;
    for (int c = 0; c < C; ++c) s += __expf(p[c] - m);
    float ls = __logf(s);
    for (int c = 0; c < C; ++c) out[(long)n * C + c] = p[c] - m - ls;
}

// ---------------- host orchestration ----------------
extern "C" void kernel_launch(void* const* d_in, const int* in_sizes, int n_in,
                              void* d_out, int out_size, void* d_ws, size_t ws_size,
                              hipStream_t stream) {
    const float* x       = (const float*)d_in[0];
    const int*   eidx    = (const int*)  d_in[1];
    const int*   perm    = (const int*)  d_in[2];
    const float* W_feat  = (const float*)d_in[3];
    const float* b_feat  = (const float*)d_in[4];
    const float* W_convs = (const float*)d_in[5];
    const float* b_convs = (const float*)d_in[6];
    const float* W_ea    = (const float*)d_in[7];
    const float* b_ea    = (const float*)d_in[8];
    const float* W_na    = (const float*)d_in[9];
    const float* b_na    = (const float*)d_in[10];
    const float* W_ctx   = (const float*)d_in[11];
    const float* b_ctx   = (const float*)d_in[12];
    const float* W_obj   = (const float*)d_in[13];
    const float* b_obj   = (const float*)d_in[14];
    const float* W_fc1   = (const float*)d_in[15];
    const float* b_fc1   = (const float*)d_in[16];
    const float* W_fc2   = (const float*)d_in[17];
    const float* b_fc2   = (const float*)d_in[18];

    const int N = in_sizes[0] / H;
    const int E = in_sizes[1] / 2;
    const int* rows = eidx;
    const int* cols = eidx + E;
    const long NH = (long)N * H;

    // ---- bump-allocate scratch (~215 MB) ----
    char* p = (char*)d_ws;
    auto alloc = [&](size_t bytes) -> char* {
        char* r = p; p += (bytes + 255) & ~(size_t)255; return r;
    };
    float* h     = (float*)alloc(NH * 4);
    float* t     = (float*)alloc(NH * 4);
    float* agg   = (float*)alloc(NH * 4);
    float* xc    = (float*)alloc(NH * 4);
    float* xo    = (float*)alloc(NH * 4);
    float* attb  = (float*)alloc((long)N * 16 * 4);   // att projections / fc2 logits
    float* ea    = (float*)alloc((long)E * 2 * 4);
    float* deg_u = (float*)alloc((long)N * 4);
    float* deg_c = (float*)alloc((long)N * 4);
    float* deg_o = (float*)alloc((long)N * 4);
    float* sA    = (float*)alloc(256 * 4);            // sum[128] | sumsq[128]
    float* sB    = (float*)alloc(256 * 4);
    float* sC    = (float*)alloc(256 * 4);
    float* mu    = (float*)alloc(128 * 4);
    float* rstd  = (float*)alloc(128 * 4);
    float* biasp = (float*)alloc(128 * 4);
    unsigned short* Abg = (unsigned short*)alloc(NH * 2);  // generic input stream
    unsigned short* Abh = (unsigned short*)alloc(NH * 2);  // h stream
    unsigned short* Abc = (unsigned short*)alloc(NH * 2);  // xc stream
    unsigned short* Abo = (unsigned short*)alloc(NH * 2);  // xo stream
    unsigned short* WT  = (unsigned short*)alloc(128 * H * 2);

    auto blocks = [](long n, int bs) { return dim3((unsigned)((n + bs - 1) / bs)); };
    auto zero = [&](float* s) { k_fill<<<1, 256, 0, stream>>>(s, 256, 0.f); };
    auto fin  = [&](float* s) { k_stats_fin<<<1, 128, 0, stream>>>(s, mu, rstd, N); };
    auto gemm = [&](const unsigned short* Ain, float* outp, unsigned short* bfOut,
                    float* statsOut, int ld, int tilesN, int relu) {
        long waves = (long)((N + 15) / 16) * ((tilesN + 1) / 2);
        k_gemm<<<blocks(waves, 8), 256, 0, stream>>>(Ain, WT, biasp, outp, bfOut, statsOut,
                                                     N, ld, tilesN, relu);
    };

    // ---- feat: h = relu(bn(x) @ W_feat + b_feat), fused bf16 + stats(h) ----
    zero(sA);
    k_colstats<<<blocks(N, 256), 128, 0, stream>>>(x, nullptr, nullptr, sA, N, 256);
    fin(sA);
    k_fold<<<1, 128, 0, stream>>>(W_feat, b_feat, mu, rstd, WT, biasp, H, H);
    k_cvt<<<blocks(NH, 256), 256, 0, stream>>>(x, nullptr, nullptr, Abg, N);
    zero(sA);
    gemm(Abg, h, Abh, sA, H, 8, 1);

    // ---- unweighted symmetric degree (shared by 3 convs) ----
    k_fill<<<blocks(N, 256), 256, 0, stream>>>(deg_u, N, 1.f);   // self loop
    k_deg_edges<<<blocks(E, 256), 256, 0, stream>>>(rows, deg_u, E);
    k_rsqrt<<<blocks(N, 256), 256, 0, stream>>>(deg_u, N);

    // ---- 3x GCN conv: h = relu( propagate( bn(h) @ Wc ) + bc ) ----
    for (int i = 0; i < 3; ++i) {
        fin(sA);
        k_fold<<<1, 128, 0, stream>>>(W_convs + (long)i * H * H, nullptr, mu, rstd, WT, biasp, H, H);
        gemm(Abh, t, nullptr, nullptr, H, 8, 0);
        k_selfloop<<<blocks(NH, 256), 256, 0, stream>>>(t, deg_u, agg, N);
        k_scatter<<<blocks((long)E * 32, 256), 256, 0, stream>>>(t, agg, rows, cols, deg_u, nullptr, 0, E);
        zero(sA);
        k_bias_relu<<<blocks(N, 256), 128, 0, stream>>>(agg, b_convs + (long)i * H, h, Abh, sA, N, 256);
    }

    // ---- attention: one padded 16-col WMMA GEMM for [u | v | node-att] ----
    k_fold_att<<<blocks(32 * H, 256), 256, 0, stream>>>(W_ea, W_na, b_na, WT, biasp);
    gemm(Abh, attb, nullptr, nullptr, 16, 1, 0);
    k_fill<<<blocks(N, 256), 256, 0, stream>>>(deg_c, N, 1.f);
    k_fill<<<blocks(N, 256), 256, 0, stream>>>(deg_o, N, 1.f);
    k_edge_att<<<blocks(E, 256), 256, 0, stream>>>(attb, rows, cols, b_ea, ea, deg_c, deg_o, E);
    k_rsqrt<<<blocks(N, 256), 256, 0, stream>>>(deg_c, N);
    k_rsqrt<<<blocks(N, 256), 256, 0, stream>>>(deg_o, N);
    zero(sA); zero(sB);
    k_node_att<<<blocks(N, 256), 128, 0, stream>>>(attb, h, xc, xo, Abc, Abo, sA, sB, N, 256);

    // ---- weighted convs (ctx / obj) ----
    struct { const float *W, *b, *dinv; float* io; unsigned short* ab; float* st; int comp; } wc[2] = {
        { W_ctx, b_ctx, deg_c, xc, Abc, sA, 0 },
        { W_obj, b_obj, deg_o, xo, Abo, sB, 1 },
    };
    for (int i = 0; i < 2; ++i) {
        fin(wc[i].st);
        k_fold<<<1, 128, 0, stream>>>(wc[i].W, nullptr, mu, rstd, WT, biasp, H, H);
        gemm(wc[i].ab, t, nullptr, nullptr, H, 8, 0);
        k_selfloop<<<blocks(NH, 256), 256, 0, stream>>>(t, wc[i].dinv, agg, N);
        k_scatter<<<blocks((long)E * 32, 256), 256, 0, stream>>>(t, agg, rows, cols, wc[i].dinv, ea, wc[i].comp, E);
        zero(wc[i].st);
        k_bias_relu<<<blocks(N, 256), 128, 0, stream>>>(agg, wc[i].b, wc[i].io, wc[i].ab, wc[i].st, N, 256);
    }

    // ---- readout heads ----
    float* out = (float*)d_out;
    auto readout = [&](const unsigned short* srcAb, float* srcStats, unsigned short* bf1,
                       const float* W1, const float* b1,
                       const float* W2, const float* b2, float* outp) {
        fin(srcStats);
        k_fold<<<1, 128, 0, stream>>>(W1, b1, mu, rstd, WT, biasp, H, H);
        zero(sC);
        gemm(srcAb, t, bf1, sC, H, 8, 1);        // relu(bn(src)@W1 + b1) + bf16 + stats
        fin(sC);
        k_fold<<<1, 128, 0, stream>>>(W2, b2, mu, rstd, WT, biasp, 10, 32);
        gemm(bf1, attb, nullptr, nullptr, 16, 1, 0);  // bn(relu)@W2 + b2 (padded)
        k_logsoftmax<<<blocks(N, 256), 256, 0, stream>>>(attb, outp, N, 10);
    };
    readout(Abc, sA, Abg, W_fc1,               b_fc1,         W_fc2,                b_fc2,          out);
    readout(Abo, sB, Abg, W_fc1 + (long)H * H, b_fc1 + H,     W_fc2 + (long)H * 10, b_fc2 + 10,     out + (long)N * 10);
    // z = xc[perm] + xo (standalone stats + cvt, then same head)
    zero(sA);
    k_colstats<<<blocks(N, 256), 128, 0, stream>>>(xc, xo, perm, sA, N, 256);
    k_cvt<<<blocks(NH, 256), 256, 0, stream>>>(xc, xo, perm, Abg, N);
    readout(Abg, sA, Abh, W_fc1 + 2L * H * H, b_fc1 + 2 * H, W_fc2 + 2L * H * 10, b_fc2 + 2 * 10, out + 2L * N * 10);
}
